// PilotRoutedMoE_90529320665482
// MI455X (gfx1250) — matrix operand order repaired and checked by reading, not compiled, not run on hardware
//
#include <hip/hip_runtime.h>

// ---------------------------------------------------------------------------
// PilotRoutedMoE on MI455X (gfx1250): bf16 WMMA pipeline, fused MoE in LDS.
// ---------------------------------------------------------------------------

typedef __bf16 bf16_t;
typedef __attribute__((ext_vector_type(16))) __bf16 v16bf;
typedef __attribute__((ext_vector_type(8)))  __bf16 v8bf;
typedef __attribute__((ext_vector_type(8)))  float  v8f;
typedef __attribute__((ext_vector_type(4)))  int    v4i;

#define B_TOK 16384
#define HID   1024
#define NEXP  8
#define P2    64      // 2P
#define MOE_M 64
#define LDSK  1032    // padded bf16 row stride for LDS tiles (bank-conflict free)

// Optional gfx1250 async global->LDS copy path (ASYNCcnt-tracked).
#if defined(__has_builtin)
#  if __has_builtin(__builtin_amdgcn_global_load_async_to_lds_b128) && \
      __has_builtin(__builtin_amdgcn_s_wait_asynccnt)
#    define USE_ASYNC_LDS 1
#  endif
#endif
#ifndef USE_ASYNC_LDS
#  define USE_ASYNC_LDS 0
#endif

#if USE_ASYNC_LDS
// per hipcc diagnostic: arg0 is int4 in addrspace(1) ("__device__"), arg1 LDS.
typedef v4i __attribute__((address_space(1))) as1_v4i;
typedef v4i __attribute__((address_space(3))) as3_v4i;
#endif

// -------------------------- small helpers ----------------------------------

__device__ __forceinline__ v8f zero8() {
  v8f z;
#pragma unroll
  for (int i = 0; i < 8; ++i) z[i] = 0.f;
  return z;
}

__device__ __forceinline__ v8f wmma_bf16(v16bf a, v16bf b, v8f c) {
  // D = A(16x32) * B(32x16) + C(16x16 f32)
  return __builtin_amdgcn_wmma_f32_16x16x32_bf16(false, a, false, b,
                                                 (short)0, c, false, false);
}

// A fragment (16x32 bf16, row-major [M][K] source):
// lanes 0-15: M=lane, K 0..7 and 16..23 ; lanes 16-31: M=lane-16, K 8..15 and 24..31
__device__ __forceinline__ v16bf load_a_bf16(const bf16_t* base, int ld) {
  const int lane = threadIdx.x & 31;
  const int m    = lane & 15;
  const int ks   = (lane < 16) ? 0 : 8;
  const bf16_t* p = base + (size_t)m * ld + ks;
  v8bf lo = *(const v8bf*)(p);
  v8bf hi = *(const v8bf*)(p + 16);
  v16bf r;
#pragma unroll
  for (int i = 0; i < 8; ++i) { r[i] = lo[i]; r[i + 8] = hi[i]; }
  return r;
}

// Same A fragment but converting from fp32 source on the fly.
__device__ __forceinline__ v16bf load_a_f32(const float* base, int ld) {
  const int lane = threadIdx.x & 31;
  const int m    = lane & 15;
  const int ks   = (lane < 16) ? 0 : 8;
  const float* p = base + (size_t)m * ld + ks;
  v16bf r;
#pragma unroll
  for (int i = 0; i < 8; ++i) {
    r[i]     = (bf16_t)p[i];
    r[i + 8] = (bf16_t)p[i + 16];
  }
  return r;
}

// B fragment (32x16 bf16) from [N][K]-major storage:
// lanes 0-15: N=lane, K 0..7 and 16..23 ; lanes 16-31: N=lane-16, K 8..15 and 24..31
__device__ __forceinline__ v16bf load_b_bf16(const bf16_t* base, int ld) {
  const int lane = threadIdx.x & 31;
  const int n    = lane & 15;
  const int ks   = (lane < 16) ? 0 : 8;
  const bf16_t* p = base + (size_t)n * ld + ks;
  v8bf lo = *(const v8bf*)(p);
  v8bf hi = *(const v8bf*)(p + 16);
  v16bf r;
#pragma unroll
  for (int i = 0; i < 8; ++i) { r[i] = lo[i]; r[i + 8] = hi[i]; }
  return r;
}

__device__ __forceinline__ float wave_sum(float v) {
#pragma unroll
  for (int o = 16; o > 0; o >>= 1) v += __shfl_xor(v, o, 32);
  return v;
}
__device__ __forceinline__ float half_sum(float v) {  // sum within 16-lane half
#pragma unroll
  for (int o = 8; o > 0; o >>= 1) v += __shfl_xor(v, o, 32);
  return v;
}

// -------------------------- prep kernels -----------------------------------

// LayerNorm row -> bf16.  grid = B_TOK, block = 256.
__global__ void __launch_bounds__(256)
k_layernorm(const float* __restrict__ x, const float* __restrict__ g,
            const float* __restrict__ be, bf16_t* __restrict__ xb) {
  const int row = blockIdx.x;
  const float* xr = x + (size_t)row * HID;
  float v[4], s = 0.f, sq = 0.f;
#pragma unroll
  for (int i = 0; i < 4; ++i) {
    v[i] = xr[threadIdx.x + i * 256];
    s += v[i]; sq += v[i] * v[i];
  }
  __shared__ float red[2][8];
  s = wave_sum(s); sq = wave_sum(sq);
  if ((threadIdx.x & 31) == 0) {
    red[0][threadIdx.x >> 5] = s;
    red[1][threadIdx.x >> 5] = sq;
  }
  __syncthreads();
  float ts = 0.f, tq = 0.f;
#pragma unroll
  for (int i = 0; i < 8; ++i) { ts += red[0][i]; tq += red[1][i]; }
  const float mean = ts * (1.f / HID);
  const float var  = tq * (1.f / HID) - mean * mean;
  const float rstd = rsqrtf(var + 1e-5f);
  bf16_t* o = xb + (size_t)row * HID;
#pragma unroll
  for (int i = 0; i < 4; ++i) {
    const int c = threadIdx.x + i * 256;
    o[c] = (bf16_t)((v[i] - mean) * rstd * g[c] + be[c]);
  }
}

// Tiled transpose + fp32->bf16:  in [batch][K][N] f32  ->  out [batch][N][K] bf16
__global__ void __launch_bounds__(256)
k_transpose_cvt(const float* __restrict__ in, bf16_t* __restrict__ out,
                int K, int N, long long inBS, long long outBS) {
  __shared__ float tile[32][33];
  const int b  = blockIdx.z;
  const int n0 = blockIdx.x * 32, k0 = blockIdx.y * 32;
  const int tx = threadIdx.x & 31, ty = threadIdx.x >> 5;  // 32 x 8
  const float* ip = in  + (size_t)b * inBS;
  bf16_t*      op = out + (size_t)b * outBS;
#pragma unroll
  for (int i = 0; i < 4; ++i) {
    const int k = k0 + ty + i * 8;
    if (k < K && (n0 + tx) < N) tile[ty + i * 8][tx] = ip[(size_t)k * N + n0 + tx];
  }
  __syncthreads();
#pragma unroll
  for (int i = 0; i < 4; ++i) {
    const int n = n0 + ty + i * 8;
    const int k = k0 + tx;
    if (n < N && k < K) op[(size_t)n * K + k] = (bf16_t)tile[tx][ty + i * 8];
  }
}

// L2-normalize pilot rows -> bf16 [E*C][H].  grid = 32, block = 256.
__global__ void __launch_bounds__(256)
k_pilot_norm(const float* __restrict__ pe, bf16_t* __restrict__ pn) {
  const int row = blockIdx.x;
  const float* pr = pe + (size_t)row * HID;
  float v[4], sq = 0.f;
#pragma unroll
  for (int i = 0; i < 4; ++i) {
    v[i] = pr[threadIdx.x + i * 256];
    sq += v[i] * v[i];
  }
  __shared__ float red[8];
  sq = wave_sum(sq);
  if ((threadIdx.x & 31) == 0) red[threadIdx.x >> 5] = sq;
  __syncthreads();
  float t = 0.f;
#pragma unroll
  for (int i = 0; i < 8; ++i) t += red[i];
  const float sc = 1.f / fmaxf(sqrtf(t), 1e-12f);
  bf16_t* o = pn + (size_t)row * HID;
#pragma unroll
  for (int i = 0; i < 4; ++i) o[threadIdx.x + i * 256] = (bf16_t)(v[i] * sc);
}

// ------------------- qproj GEMM + fused row l2norm --------------------------
// q = l2norm( [x|query] @ Wq + b ).  M-block = 16 rows, 8 waves x 128 cols.
__global__ void __launch_bounds__(256)
k_qproj(const bf16_t* __restrict__ xb, const float* __restrict__ qf,
        const bf16_t* __restrict__ Wqt, const float* __restrict__ qb,
        bf16_t* __restrict__ qn) {
  const int row0  = blockIdx.x * 16;
  const int w     = threadIdx.x >> 5;
  const int lane  = threadIdx.x & 31;
  const int nl    = lane & 15;
  const int half8 = (lane < 16) ? 0 : 8;

  v8f acc[8];
#pragma unroll
  for (int t = 0; t < 8; ++t) acc[t] = zero8();

  for (int kt = 0; kt < 64; ++kt) {
    const int k0 = kt * 32;
    v16bf a;
    if (k0 < HID) a = load_a_bf16(xb + (size_t)row0 * HID + k0, HID);
    else          a = load_a_f32 (qf + (size_t)row0 * HID + (k0 - HID), HID);
    // batch all 8 B fragments first so the loads form one clause,
    // then drain into WMMAs with overlapped waits.
    v16bf b[8];
#pragma unroll
    for (int t = 0; t < 8; ++t)
      b[t] = load_b_bf16(Wqt + (size_t)(w * 128 + t * 16) * 2048 + k0, 2048);
#pragma unroll
    for (int t = 0; t < 8; ++t)
      acc[t] = wmma_bf16(a, b[t], acc[t]);
  }

  // bias + per-row sum of squares (deterministic two-level reduction)
  float sq[8];
#pragma unroll
  for (int r = 0; r < 8; ++r) sq[r] = 0.f;
#pragma unroll
  for (int t = 0; t < 8; ++t) {
#pragma unroll
    for (int r = 0; r < 8; ++r) {
      const float v = acc[t][r] + qb[w * 128 + t * 16 + nl];
      acc[t][r] = v;
      sq[r] += v * v;
    }
  }
  __shared__ float wss[8][16];   // [wave][row-in-block]
#pragma unroll
  for (int r = 0; r < 8; ++r) {
    const float h = half_sum(sq[r]);
    if (nl == r) wss[w][r + half8] = h;
  }
  __syncthreads();
  float rs[8];
#pragma unroll
  for (int r = 0; r < 8; ++r) {
    float tot = 0.f;
#pragma unroll
    for (int ww = 0; ww < 8; ++ww) tot += wss[ww][r + half8];
    rs[r] = 1.f / fmaxf(sqrtf(tot), 1e-12f);
  }
#pragma unroll
  for (int t = 0; t < 8; ++t) {
#pragma unroll
    for (int r = 0; r < 8; ++r) {
      const int m = r + half8;
      qn[(size_t)(row0 + m) * HID + w * 128 + t * 16 + nl] =
          (bf16_t)(acc[t][r] * rs[r]);
    }
  }
}

// ------------------- sims GEMM + softmax/top-2 router -----------------------
// M-block = 64 rows, N = 32 (= E*C).  8 waves: (m 0..3) x (n 0..1).
__global__ void __launch_bounds__(256)
k_router(const bf16_t* __restrict__ qn, const bf16_t* __restrict__ pn,
         float* __restrict__ cw) {
  const int row0  = blockIdx.x * 64;
  const int w     = threadIdx.x >> 5;
  const int lane  = threadIdx.x & 31;
  const int nl    = lane & 15;
  const int half8 = (lane < 16) ? 0 : 8;
  const int mt    = w >> 1, nt = w & 1;

  v8f acc = zero8();
  for (int kt = 0; kt < 32; ++kt) {
    const int k0 = kt * 32;
    v16bf a = load_a_bf16(qn + (size_t)(row0 + mt * 16) * HID + k0, HID);
    v16bf b = load_b_bf16(pn + (size_t)(nt * 16) * HID + k0, HID);
    acc = wmma_bf16(a, b, acc);
  }
  __shared__ float sm[64][33];
#pragma unroll
  for (int r = 0; r < 8; ++r)
    sm[mt * 16 + r + half8][nt * 16 + nl] = acc[r];
  __syncthreads();

  if (threadIdx.x < 64) {
    const int t = threadIdx.x;
    float sc[NEXP];
#pragma unroll
    for (int e = 0; e < NEXP; ++e)
      sc[e] = (sm[t][4*e] + sm[t][4*e+1] + sm[t][4*e+2] + sm[t][4*e+3]) * 0.25f
              * 10.0f;  // / TEMP
    float mx = sc[0];
#pragma unroll
    for (int e = 1; e < NEXP; ++e) mx = fmaxf(mx, sc[e]);
    float p[NEXP], Z = 0.f;
#pragma unroll
    for (int e = 0; e < NEXP; ++e) { p[e] = expf(sc[e] - mx); Z += p[e]; }
    const float iZ = 1.f / Z;
#pragma unroll
    for (int e = 0; e < NEXP; ++e) p[e] *= iZ;
    int i1 = 0; float p1 = p[0];
#pragma unroll
    for (int e = 1; e < NEXP; ++e) if (p[e] > p1) { p1 = p[e]; i1 = e; }
    int i2 = -1; float p2 = -1.f;
#pragma unroll
    for (int e = 0; e < NEXP; ++e)
      if (e != i1 && p[e] > p2) { p2 = p[e]; i2 = e; }
    const float inv = 1.f / (p1 + p2 + 1e-6f);
    float* o = cw + (size_t)(row0 + t) * NEXP;
#pragma unroll
    for (int e = 0; e < NEXP; ++e) o[e] = 0.f;
    o[i1] = p1 * inv;
    o[i2] = p2 * inv;
  }
}

// ------------------- fused MoE: experts + shared, LDS-resident X and H ------
// Per block: 64 tokens. e = 0..7 routed experts, e = 8 shared expert (weight 1).
// X block staged once in LDS (async copy when available); H tile lives in LDS;
// 264 KB of the 320 KB WGP LDS in use.
__global__ void __launch_bounds__(256)
k_moe(const bf16_t* __restrict__ xb,
      const bf16_t* __restrict__ W1t, const float* __restrict__ eb1,
      const bf16_t* __restrict__ W2t, const float* __restrict__ eb2,
      const bf16_t* __restrict__ sW1t, const float* __restrict__ sb1,
      const bf16_t* __restrict__ sW2t, const float* __restrict__ sb2,
      const float* __restrict__ cw, float* __restrict__ comb) {
  __shared__ __align__(16) bf16_t Xs[MOE_M * LDSK];   // 132 KB: x block
  __shared__ __align__(16) bf16_t Hs[MOE_M * LDSK];   // 132 KB: relu(x@W1+b1)

  const int row0  = blockIdx.x * MOE_M;
  const int w     = threadIdx.x >> 5;
  const int lane  = threadIdx.x & 31;
  const int nl    = lane & 15;
  const int half8 = (lane < 16) ? 0 : 8;
  const int ms  = w & 3;               // 16-row strip owned by this wave
  const int nh  = w >> 2;              // GEMM1: which 512-col half
  const int nt2 = (w >> 2) * 2;        // GEMM2: first of two 16-col tiles

  // ---- stage x block into LDS (row stride re-padded to LDSK)
#if USE_ASYNC_LDS
  for (int i = threadIdx.x; i < MOE_M * (HID / 8); i += 256) {
    const int r = i >> 7, c = i & 127;   // c: 16-byte chunk within row
    __builtin_amdgcn_global_load_async_to_lds_b128(
        (as1_v4i*)(xb + (size_t)(row0 + r) * HID + c * 8),
        (as3_v4i*)&Xs[(size_t)r * LDSK + c * 8], 0, 0);
  }
  __builtin_amdgcn_s_wait_asynccnt(0);
#else
  for (int i = threadIdx.x; i < MOE_M * (HID / 8); i += 256) {
    const int r = i >> 7, c = i & 127;
    *(v8bf*)&Xs[(size_t)r * LDSK + c * 8] =
        *(const v8bf*)(xb + (size_t)(row0 + r) * HID + c * 8);
  }
#endif
  __syncthreads();

  float racc[2][8];
#pragma unroll
  for (int j = 0; j < 2; ++j)
#pragma unroll
    for (int r = 0; r < 8; ++r) racc[j][r] = 0.f;

  for (int e = 0; e < 9; ++e) {
    const bf16_t* w1 = (e < 8) ? (W1t + (size_t)e * HID * HID) : sW1t;
    const float*  b1 = (e < 8) ? (eb1 + e * HID)               : sb1;
    const bf16_t* w2 = (e < 8) ? (W2t + (size_t)e * P2 * HID)  : sW2t;
    const float*  b2 = (e < 8) ? (eb2 + e * P2)                : sb2;
    if (e) __syncthreads();            // previous GEMM2 done reading Hs

    // ---- GEMM1: Hs = relu(x_blk @ w1 + b1), bf16, [64][1024] in LDS.
    //      B double-buffered across k iterations; A from LDS.
    for (int ncb = 0; ncb < 8; ++ncb) {
      const int n0 = nh * 512 + ncb * 64;
      v8f acc[4];
#pragma unroll
      for (int j = 0; j < 4; ++j) acc[j] = zero8();
      v16bf bcur[4];
#pragma unroll
      for (int j = 0; j < 4; ++j)
        bcur[j] = load_b_bf16(w1 + (size_t)(n0 + j * 16) * HID, HID);
      for (int kt = 0; kt < 32; ++kt) {
        const int k0 = kt * 32;
        const int kn = (kt < 31) ? k0 + 32 : k0;   // safe prefetch address
        v16bf bnext[4];
#pragma unroll
        for (int j = 0; j < 4; ++j)
          bnext[j] = load_b_bf16(w1 + (size_t)(n0 + j * 16) * HID + kn, HID);
        v16bf a = load_a_bf16(Xs + (size_t)(ms * 16) * LDSK + k0, LDSK);
#pragma unroll
        for (int j = 0; j < 4; ++j) acc[j] = wmma_bf16(a, bcur[j], acc[j]);
#pragma unroll
        for (int j = 0; j < 4; ++j) bcur[j] = bnext[j];
      }
#pragma unroll
      for (int j = 0; j < 4; ++j) {
        const int n = n0 + j * 16 + nl;
#pragma unroll
        for (int r = 0; r < 8; ++r) {
          float v = acc[j][r] + b1[n];
          v = fmaxf(v, 0.f);
          Hs[(size_t)(ms * 16 + r + half8) * LDSK + n] = (bf16_t)v;
        }
      }
    }
    __syncthreads();

    // ---- GEMM2: eout = Hs @ w2 + b2 ; weighted accumulate into racc
    float cwv[8];
#pragma unroll
    for (int r = 0; r < 8; ++r) cwv[r] = 0.f;
    if (e < 8) {
#pragma unroll
      for (int r = 0; r < 8; ++r)
        cwv[r] = cw[(size_t)(row0 + ms * 16 + r + half8) * NEXP + e];
    }
    v8f acc2[2];
    acc2[0] = zero8(); acc2[1] = zero8();
    v16bf b2cur[2];
#pragma unroll
    for (int j = 0; j < 2; ++j)
      b2cur[j] = load_b_bf16(w2 + (size_t)((nt2 + j) * 16) * HID, HID);
    for (int kt = 0; kt < 32; ++kt) {
      const int k0 = kt * 32;
      const int kn = (kt < 31) ? k0 + 32 : k0;
      v16bf b2next[2];
#pragma unroll
      for (int j = 0; j < 2; ++j)
        b2next[j] = load_b_bf16(w2 + (size_t)((nt2 + j) * 16) * HID + kn, HID);
      v16bf a = load_a_bf16(Hs + (size_t)(ms * 16) * LDSK + k0, LDSK);
#pragma unroll
      for (int j = 0; j < 2; ++j) acc2[j] = wmma_bf16(a, b2cur[j], acc2[j]);
#pragma unroll
      for (int j = 0; j < 2; ++j) b2cur[j] = b2next[j];
    }
#pragma unroll
    for (int j = 0; j < 2; ++j) {
      const int n = (nt2 + j) * 16 + nl;
#pragma unroll
      for (int r = 0; r < 8; ++r) {
        const float v = acc2[j][r] + b2[n];
        if (e < 8)
          racc[j][r] += cwv[r] * v;
        else
          comb[(size_t)(row0 + ms * 16 + r + half8) * 128 + 64 + n] = v;  // shared
      }
    }
  }

  // routed half of "combined"
#pragma unroll
  for (int j = 0; j < 2; ++j) {
    const int n = (nt2 + j) * 16 + nl;
#pragma unroll
    for (int r = 0; r < 8; ++r)
      comb[(size_t)(row0 + ms * 16 + r + half8) * 128 + n] = racc[j][r];
  }
}

// ------------------- gate + sigmoid ----------------------------------------
// out[b][n] = sigmoid( combined[b][:] . gate_W[:,n] + gate_b[n] ), K=128, N=64
__global__ void __launch_bounds__(256)
k_gate(const float* __restrict__ comb, const float* __restrict__ gW,
       const float* __restrict__ gb, float* __restrict__ out) {
  __shared__ float Wl[128 * 64];
  __shared__ float Cr[4][128];
  for (int i = threadIdx.x; i < 128 * 64; i += 256) Wl[i] = gW[i];
  const int row0 = blockIdx.x * 4;
  for (int i = threadIdx.x; i < 4 * 128; i += 256)
    Cr[i >> 7][i & 127] = comb[(size_t)row0 * 128 + i];
  __syncthreads();
  const int r = threadIdx.x >> 6;
  const int n = threadIdx.x & 63;
  float acc = gb[n];
#pragma unroll 8
  for (int k = 0; k < 128; ++k) acc += Cr[r][k] * Wl[k * 64 + n];
  out[(size_t)(row0 + r) * 64 + n] = 1.f / (1.f + expf(-acc));
}

// ---------------------------------------------------------------------------

extern "C" void kernel_launch(void* const* d_in, const int* in_sizes, int n_in,
                              void* d_out, int out_size, void* d_ws, size_t ws_size,
                              hipStream_t stream) {
  const float* mm    = (const float*)d_in[0];
  const float* qf    = (const float*)d_in[1];
  const float* g     = (const float*)d_in[2];
  const float* be    = (const float*)d_in[3];
  const float* pilot = (const float*)d_in[4];
  const float* qW    = (const float*)d_in[5];
  const float* qb    = (const float*)d_in[6];
  const float* eW1   = (const float*)d_in[7];
  const float* eb1   = (const float*)d_in[8];
  const float* eW2   = (const float*)d_in[9];
  const float* eb2   = (const float*)d_in[10];
  const float* sW1   = (const float*)d_in[11];
  const float* sb1   = (const float*)d_in[12];
  const float* sW2   = (const float*)d_in[13];
  const float* sb2   = (const float*)d_in[14];
  const float* gW    = (const float*)d_in[15];
  const float* gb    = (const float*)d_in[16];
  float* out = (float*)d_out;

  size_t off = 0;
  auto alloc = [&](size_t bytes) -> void* {
    void* p = (char*)d_ws + off;
    off += (bytes + 255) & ~(size_t)255;
    return p;
  };
  bf16_t* xb   = (bf16_t*)alloc((size_t)B_TOK * HID * 2);
  bf16_t* qn   = (bf16_t*)alloc((size_t)B_TOK * HID * 2);
  bf16_t* Wqt  = (bf16_t*)alloc((size_t)HID * 2 * HID * 2);
  bf16_t* W1t  = (bf16_t*)alloc((size_t)NEXP * HID * HID * 2);
  bf16_t* W2t  = (bf16_t*)alloc((size_t)NEXP * P2 * HID * 2);
  bf16_t* sW1t = (bf16_t*)alloc((size_t)HID * HID * 2);
  bf16_t* sW2t = (bf16_t*)alloc((size_t)P2 * HID * 2);
  bf16_t* pn   = (bf16_t*)alloc((size_t)32 * HID * 2);
  float*  cwp  = (float*)alloc((size_t)B_TOK * NEXP * 4);
  float*  comb = (float*)alloc((size_t)B_TOK * 128 * 4);
  (void)ws_size; (void)n_in; (void)in_sizes; (void)out_size;

  const dim3 tb(256);
  const dim3 tt(256);

  k_layernorm<<<B_TOK, tb, 0, stream>>>(mm, g, be, xb);

  // weight transposes: in [K][N] f32 -> out [N][K] bf16
  k_transpose_cvt<<<dim3(HID / 32, 2 * HID / 32, 1), tt, 0, stream>>>(
      qW, Wqt, 2 * HID, HID, 0, 0);
  k_transpose_cvt<<<dim3(HID / 32, HID / 32, NEXP), tt, 0, stream>>>(
      eW1, W1t, HID, HID, (long long)HID * HID, (long long)HID * HID);
  k_transpose_cvt<<<dim3(P2 / 32, HID / 32, NEXP), tt, 0, stream>>>(
      eW2, W2t, HID, P2, (long long)HID * P2, (long long)P2 * HID);
  k_transpose_cvt<<<dim3(HID / 32, HID / 32, 1), tt, 0, stream>>>(
      sW1, sW1t, HID, HID, 0, 0);
  k_transpose_cvt<<<dim3(P2 / 32, HID / 32, 1), tt, 0, stream>>>(
      sW2, sW2t, HID, P2, 0, 0);

  k_pilot_norm<<<32, tb, 0, stream>>>(pilot, pn);

  k_qproj<<<B_TOK / 16, tb, 0, stream>>>(xb, qf, Wqt, qb, qn);
  k_router<<<B_TOK / 64, tb, 0, stream>>>(qn, pn, cwp);
  k_moe<<<B_TOK / MOE_M, tb, 0, stream>>>(xb, W1t, eb1, W2t, eb2,
                                          sW1t, sb1, sW2t, sb2, cwp, comb);
  k_gate<<<B_TOK / 4, tb, 0, stream>>>(comb, gW, gb, out);
}
